// NaiveLSTM_66915590472230
// MI455X (gfx1250) — compile-verified, matrix operand order checked by reading, not run on hardware
//
#include <hip/hip_runtime.h>
#include <hip/hip_bf16.h>

// ---------------------------------------------------------------------------
// Persistent-kernel LSTM on gfx1250 (single launch, bf16 WMMA, TDM x-staging).
//   B=256, T=1024, D=256, H=256, 4H=1024
// Grid: 16 row-groups x 16 column-WGs. Each WG owns 16 batch rows (shared by
// its group) and one 16-wide h-column block. Per step:
//   waves 0-3: gate tiles (i,f,j,o), K=512 over [x_t | h_{t-1}] from LDS
//   wave  4 : y_{t-1} = h_{t-1} @ W_y + b_y  (A from LDS, B from LDS)
// Cross-WG h exchange via L2 + per-group atomic barrier (16 WGs).
// x_{t+1} staged into LDS by the Tensor Data Mover (double buffered).
// ---------------------------------------------------------------------------

#define B_   256
#define T_   1024
#define D_   256
#define H_   256
#define KC_  512      // concat K for the cell GEMM
#define LDA_ 520      // lds_A row stride in halfs (512 + 8 pad, conflict-free)

typedef __attribute__((ext_vector_type(16))) __bf16         v16bf;
typedef __attribute__((ext_vector_type(16))) unsigned short v16u;
typedef __attribute__((ext_vector_type(8)))  unsigned short v8u;
typedef __attribute__((ext_vector_type(4)))  unsigned short v4us;
typedef __attribute__((ext_vector_type(8)))  float          v8f;
typedef __attribute__((ext_vector_type(4)))  float          v4f;
typedef __attribute__((ext_vector_type(4)))  unsigned int   v4ui;
typedef __attribute__((ext_vector_type(8)))  int            v8i;
typedef __attribute__((ext_vector_type(4)))  int            v4i;

__device__ __forceinline__ unsigned short f2bfu(float f) {
  union { float f; unsigned u; } v; v.f = f;
  unsigned r = (v.u + 0x7FFFu + ((v.u >> 16) & 1u)) >> 16;
  return (unsigned short)r;
}
__device__ __forceinline__ float bfu2f(unsigned short s) {
  union { unsigned u; float f; } o; o.u = ((unsigned)s) << 16;
  return o.f;
}
__device__ __forceinline__ float sigmoidf_(float x) {
  return 1.0f / (1.0f + __expf(-x));
}
__device__ __forceinline__ v16bf make_frag(v8u lo, v8u hi) {
  v16u u;
#pragma unroll
  for (int q = 0; q < 8; ++q) { u[q] = lo[q]; u[q + 8] = hi[q]; }
  return __builtin_bit_cast(v16bf, u);
}

// TDM: load a 16-row x 256-fp32 tile of x (row stride T*D floats) into LDS.
// D# group0: count=1, lds_addr, 57b global addr, type=2.
// D# group1: data_size=4B(code 2), tensor_dim0=256, tensor_dim1=16,
//            tile_dim0=256, tile_dim1=16, tensor_dim0_stride=T*D.
// This toolchain's builtin is the 6-arg form:
//   (uint32x4 g0, int32x8 g1, int32x4 g2, int32x4 g3, int32x8 g4, i32 cpol)
__device__ __forceinline__ void tdm_load_x_tile(const float* gsrc, unsigned lds_off) {
  unsigned long long ga = (unsigned long long)(uintptr_t)gsrc;
  v4ui g0 = { 1u, lds_off, (unsigned)(ga & 0xFFFFFFFFu),
              ((unsigned)((ga >> 32) & 0x1FFFFFFull)) | (2u << 30) };
  v8i g1 = { (int)0x00020000,          // workgroup_mask=0, data_size=2 (4B)
             (int)(256u << 16),        // tensor_dim0 = 256 (lo16 in hi half)
             (int)(16u << 16),         // tensor_dim0 hi=0 | tensor_dim1 = 16
             (int)(256u << 16),        // tensor_dim1 hi=0 | tile_dim0 = 256
             (int)16,                  // tile_dim1 = 16, tile_dim2 = 0
             (int)(T_ * D_),           // tensor_dim0_stride lo32 = 262144
             0, 0 };
  v4i gz4 = { 0, 0, 0, 0 };
  v8i gz8 = { 0, 0, 0, 0, 0, 0, 0, 0 };
  __builtin_amdgcn_tensor_load_to_lds(g0, g1, gz4, gz4, gz8, 0);
}

// ---------------------------------------------------------------------------
// Weight packing into WMMA-fragment-major layout:
// lane L -> kbase=(L>=16)?8:0 ; half i -> k = kbase + (i<8 ? i : i+8); col=L%16
// ---------------------------------------------------------------------------
__global__ void lstm_pack_wc(const float* __restrict__ Wih,
                             const float* __restrict__ Whh,
                             unsigned short* __restrict__ out) {
  int idx = blockIdx.x * blockDim.x + threadIdx.x;   // 0 .. 1024*512-1
  int i    = idx & 15;
  int lane = (idx >> 4) & 31;
  int kb   = (idx >> 9) & 15;
  int nb   = idx >> 13;                               // 0..63 col block of 4H
  int kbase = (lane >= 16) ? 8 : 0;
  int k = kb * 32 + kbase + ((i < 8) ? i : (i + 8));  // 0..511
  int g = nb * 16 + (lane & 15);                      // 0..1023 gate column
  float v = (k < D_) ? Wih[g * D_ + k] : Whh[g * H_ + (k - D_)];
  out[idx] = f2bfu(v);
}

__global__ void lstm_pack_wy(const float* __restrict__ Wy,
                             unsigned short* __restrict__ out) {
  int idx = blockIdx.x * blockDim.x + threadIdx.x;   // 0 .. 256*256-1
  int i    = idx & 15;
  int lane = (idx >> 4) & 31;
  int kb   = (idx >> 9) & 7;
  int nb   = idx >> 12;                               // 0..15 d-col block
  int kbase = (lane >= 16) ? 8 : 0;
  int k = kb * 32 + kbase + ((i < 8) ? i : (i + 8));  // 0..255 (h index)
  int n = nb * 16 + (lane & 15);                      // 0..255 (d index)
  out[idx] = f2bfu(Wy[k * D_ + n]);                   // W_y row major [H, D]
}

__global__ void lstm_init_state(const float* __restrict__ h0,
                                unsigned short* __restrict__ hstate0,
                                unsigned* __restrict__ bar) {
  int idx = blockIdx.x * blockDim.x + threadIdx.x;   // 0 .. B*H-1
  hstate0[idx] = f2bfu(h0[idx & (H_ - 1)]);          // h0 broadcast over batch
  if (idx < 16) bar[idx] = 0u;                       // group barrier counters
}

// ---------------------------------------------------------------------------
// Dynamic-LDS carve (bytes):
//   lds_wc   : 0      .. 65536   B-fragments, 4 gate tiles [g][kb][ch][lane][8]
//   lds_wy   : 65536  .. 73728   B-fragments, y tile        [kb][ch][lane][8]
//   lds_A    : 73728  .. 90368   [16][520] bf16 rows of [x_t | h_{t-1}]
//   lds_x    : 90368  .. 123136  double-buffered fp32 x tile (2 x 16KB)
//   lds_gate : 123136 .. 126208  f/j/o tiles for the combine
// ---------------------------------------------------------------------------
#define SM_WC  0
#define SM_WY  65536
#define SM_A   73728
#define SM_X   90368
#define SM_G   123136
#define SM_TOT 126208

__global__ __launch_bounds__(160)
void lstm_persistent(const float* __restrict__ x,           // [B,T,D] fp32
                     const float* __restrict__ b_ih,        // [4H]
                     const float* __restrict__ c0,          // [1,H]
                     const float* __restrict__ b_y,         // [D]
                     const unsigned short* __restrict__ Wc, // packed [64][512][16]
                     const unsigned short* __restrict__ Wy, // packed [16][256][16]
                     unsigned short* __restrict__ h_state,  // [2][B*H] bf16
                     unsigned* __restrict__ bar,            // [16] group counters
                     float* __restrict__ y_out,             // [B,T,D]
                     float* __restrict__ hc_out)            // d_out + B*T*D
{
  extern __shared__ unsigned char smem[];
  unsigned short* lds_wc  = (unsigned short*)(smem + SM_WC);
  unsigned short* lds_wy  = (unsigned short*)(smem + SM_WY);
  unsigned short* lds_A   = (unsigned short*)(smem + SM_A);
  float*          lds_x   = (float*)(smem + SM_X);
  float*          lds_g   = (float*)(smem + SM_G);

  const int tid  = threadIdx.x;
  const int wave = tid >> 5;
  const int lane = tid & 31;
  const int rb   = blockIdx.x >> 4;   // row group (16 rows rb*16..)
  const int wgin = blockIdx.x & 15;   // h-column block owned by this WG
  const int kbase = (lane >= 16) ? 8 : 0;
  const int m     = lane & 15;
  const int col   = wgin * 16 + (lane & 15);

  // ---- one-time: preload this WG's weight slices into LDS ----
  for (int q = tid; q < 4096; q += 160) {           // Wc: 4 gates x 16kb x 2ch x 32ln
    int g  = q >> 10;
    int r  = q & 1023;
    int kb = r >> 6, ch = (r >> 5) & 1, l2 = r & 31;
    const unsigned short* src =
        Wc + ((size_t)(g * 16 + wgin) * 512 + kb * 32 + l2) * 16 + ch * 8;
    *(v8u*)(lds_wc + (size_t)q * 8) = *(const v8u*)src;
  }
  for (int q = tid; q < 512; q += 160) {            // Wy: 8kb x 2ch x 32ln
    int kb = q >> 6, ch = (q >> 5) & 1, l2 = q & 31;
    const unsigned short* src =
        Wy + ((size_t)wgin * 256 + kb * 32 + l2) * 16 + ch * 8;
    *(v8u*)(lds_wy + (size_t)q * 8) = *(const v8u*)src;
  }

  // ---- c tile lives in wave-0 registers for the whole sequence ----
  v8f creg;
  {
    float c0v = c0[col];
#pragma unroll
    for (int r = 0; r < 8; ++r) creg[r] = c0v;
  }
  v8f acc = {};

  const float* xbase = x + (size_t)rb * 16 * T_ * D_;   // row rb*16, t=0
  const float bias_g = b_ih[(wave < 4 ? wave : 0) * H_ + col];
  const float by     = b_y[col];

  // ---- prologue: TDM-stage x_0 into buffer 0 ----
  if (wave == 4) tdm_load_x_tile(xbase, (unsigned)(uintptr_t)(lds_x));

#pragma unroll 1
  for (int t = 0; t <= T_; ++t) {
    const int parity = t & 1;
    const unsigned short* h_prev = h_state + parity * (B_ * H_) + rb * 16 * H_;
    unsigned short*       h_next = h_state + (parity ^ 1) * (B_ * H_);

    // -- wave 4: retire x_t TDM, kick x_{t+1} TDM into the other buffer --
    if (wave == 4) {
      __builtin_amdgcn_s_wait_tensorcnt(0);
      if (t + 1 < T_) {
        tdm_load_x_tile(xbase + (size_t)(t + 1) * D_,
                        (unsigned)(uintptr_t)(lds_x + ((t + 1) & 1) * 4096));
      }
    }
    __syncthreads();   // x_t staged; h_{t-1} globally visible (group barrier)

    // -- cooperative fill of lds_A = [bf16(x_t) | h_{t-1}] --
    if (t < T_) {
      const float* xs = lds_x + parity * 4096;
      for (int q = tid; q < 1024; q += 160) {       // 16 rows x 64 quads
        int r = q >> 6;
        int kq = (q & 63) * 4;
        v4f xv = *(const v4f*)(xs + r * 256 + kq);
        v4us o; o[0] = f2bfu(xv[0]); o[1] = f2bfu(xv[1]);
                o[2] = f2bfu(xv[2]); o[3] = f2bfu(xv[3]);
        *(v4us*)(lds_A + r * LDA_ + kq) = o;
      }
    }
    for (int q = tid; q < 512; q += 160) {          // h: 16 rows x 32 v8 chunks
      int r = q >> 5;
      int kc = (q & 31) * 8;
      *(v8u*)(lds_A + r * LDA_ + 256 + kc) = *(const v8u*)(h_prev + r * H_ + kc);
    }
    __syncthreads();

    // -- waves 0-3: gate tiles, K=512, A and B both from LDS --
    if (wave < 4 && t < T_) {
      v8f c = {};
#pragma unroll 4
      for (int kb = 0; kb < 16; ++kb) {
        const unsigned short* ap = lds_A + m * LDA_ + kb * 32 + kbase;
        v8u a0 = *(const v8u*)ap;                       // ds_load_b128
        v8u a1 = *(const v8u*)(ap + 16);
        const unsigned short* bp =
            lds_wc + (((wave * 16 + kb) * 2) * 32 + lane) * 8;
        v8u b0 = *(const v8u*)bp;                       // ds_load_b128
        v8u b1 = *(const v8u*)(bp + 256);
        c = __builtin_amdgcn_wmma_f32_16x16x32_bf16(false, make_frag(a0, a1),
                                                    false, make_frag(b0, b1),
                                                    (short)0, c, false, false);
      }
#pragma unroll
      for (int r = 0; r < 8; ++r) c[r] += bias_g;
      if (wave >= 1) {
#pragma unroll
        for (int r = 0; r < 8; ++r) lds_g[(wave - 1) * 256 + r * 32 + lane] = c[r];
      }
      acc = c;
    }

    // -- wave 4: y_{t-1} = h_{t-1} @ W_y + b_y (A = lds_A h-part) --
    if (wave == 4 && t >= 1) {
      v8f c = {};
#pragma unroll
      for (int kb = 0; kb < 8; ++kb) {
        const unsigned short* ap = lds_A + m * LDA_ + 256 + kb * 32 + kbase;
        v8u a0 = *(const v8u*)ap;
        v8u a1 = *(const v8u*)(ap + 16);
        const unsigned short* bp = lds_wy + ((kb * 2) * 32 + lane) * 8;
        v8u b0 = *(const v8u*)bp;
        v8u b1 = *(const v8u*)(bp + 256);
        c = __builtin_amdgcn_wmma_f32_16x16x32_bf16(false, make_frag(a0, a1),
                                                    false, make_frag(b0, b1),
                                                    (short)0, c, false, false);
      }
#pragma unroll
      for (int r = 0; r < 8; ++r) {
        int row = rb * 16 + r + ((lane >= 16) ? 8 : 0);
        y_out[((size_t)row * T_ + (t - 1)) * D_ + col] = c[r] + by;
      }
    }
    __syncthreads();   // gate tiles published in lds_g

    // -- wave 0: combine gates, update c (registers) and h (global bf16) --
    if (wave == 0 && t < T_) {
#pragma unroll
      for (int r = 0; r < 8; ++r) {
        int row = rb * 16 + r + ((lane >= 16) ? 8 : 0);
        float iv = sigmoidf_(acc[r]);
        float fv = sigmoidf_(lds_g[0 * 256 + r * 32 + lane]);
        float jv = tanhf(lds_g[1 * 256 + r * 32 + lane]);
        float ov = sigmoidf_(lds_g[2 * 256 + r * 32 + lane]);
        float cv = fv * creg[r] + iv * jv;
        creg[r] = cv;
        h_next[(size_t)row * H_ + col] = f2bfu(ov * tanhf(cv));
      }
      __threadfence();   // release h_next to the other WGs of this group
    }
    __syncthreads();

    // -- group barrier: 16 WGs of this row-group rendezvous --
    if (t < T_) {
      if (tid == 0) {
        atomicAdd(&bar[rb], 1u);
        unsigned target = 16u * (unsigned)(t + 1);
        while (__hip_atomic_load(&bar[rb], __ATOMIC_ACQUIRE,
                                 __HIP_MEMORY_SCOPE_AGENT) < target) {
          __builtin_amdgcn_s_sleep(2);
        }
      }
      __syncthreads();
      __threadfence();   // acquire side: see peers' h_next through L2
    }

    // -- epilogue iteration (t==T): final h and c to the output tail --
    if (t == T_) {
      if (wave == 0) {
#pragma unroll
        for (int r = 0; r < 8; ++r) {
          int row = rb * 16 + r + ((lane >= 16) ? 8 : 0);
          hc_out[(size_t)B_ * H_ + (size_t)row * H_ + col] = creg[r];
        }
      }
      for (int e = tid; e < 256; e += 160) {   // h_{T-1} tile (16 x 16)
        int r = e >> 4, cl = e & 15;
        int row = rb * 16 + r, cc = wgin * 16 + cl;
        hc_out[(size_t)row * H_ + cc] = bfu2f(lds_A[r * LDA_ + 256 + cc]);
      }
    }
  }
}

// ---------------------------------------------------------------------------
extern "C" void kernel_launch(void* const* d_in, const int* in_sizes, int n_in,
                              void* d_out, int out_size, void* d_ws, size_t ws_size,
                              hipStream_t stream) {
  const float* x    = (const float*)d_in[0];
  const float* W_ih = (const float*)d_in[1];
  const float* b_ih = (const float*)d_in[2];
  const float* W_hh = (const float*)d_in[3];
  const float* h0   = (const float*)d_in[4];
  const float* c0   = (const float*)d_in[5];
  const float* W_y  = (const float*)d_in[6];
  const float* b_y  = (const float*)d_in[7];

  char* ws = (char*)d_ws;
  unsigned short* Wc     = (unsigned short*)(ws);             // 1,048,576 B
  unsigned short* Wy     = (unsigned short*)(ws + 1048576);   //   131,072 B
  unsigned short* hstate = (unsigned short*)(ws + 1179648);   //   262,144 B (x2)
  unsigned*       bar    = (unsigned*)     (ws + 1441792);    //        64 B

  float* y_out  = (float*)d_out;
  float* hc_out = y_out + (size_t)B_ * T_ * D_;

  lstm_pack_wc<<<dim3((4 * H_ * KC_) / 256), dim3(256), 0, stream>>>(W_ih, W_hh, Wc);
  lstm_pack_wy<<<dim3((H_ * D_) / 256),      dim3(256), 0, stream>>>(W_y, Wy);
  lstm_init_state<<<dim3((B_ * H_) / 256),   dim3(256), 0, stream>>>(h0, hstate, bar);

  lstm_persistent<<<dim3(256), dim3(160), SM_TOT, stream>>>(
      x, b_ih, c0, b_y, Wc, Wy, hstate, bar, y_out, hc_out);
}
